// GaussianAttention_28441273434775
// MI455X (gfx1250) — matrix operand order repaired and checked by reading, not compiled, not run on hardware
//
#include <hip/hip_runtime.h>

// ---------------------------------------------------------------------------
// Shapes: TS=256, CTS=512, MB=64, D=512, K=20
// Outputs: hiddens (256,64,512) | att_k (256,64,20) | att_w (256,64,512)
// ---------------------------------------------------------------------------

typedef __attribute__((ext_vector_type(16))) __bf16 v16bf;
typedef __attribute__((ext_vector_type(8)))  float  v8f;

// ---- WMMA fragment layouts (a_off: ISA 7.12.2 k order; b_off: contiguous k = 16h + s), element offsets within a
// 1024-byte (32 lane x 16 bf16) fragment-major tile -------------------------
__device__ __forceinline__ int a_off(int m, int k) {
  int h = (k & 15) >> 3;
  int rem = k & 7;
  int v = ((k >> 4) << 2) + (rem >> 1);
  int e = rem & 1;
  return (m + (h << 4)) * 16 + (v * 2 + e);
}
__device__ __forceinline__ int b_off(int k, int n) {
  return (n + ((k >> 4) << 4)) * 16 + (k & 15);
}

// ---------------------------------------------------------------------------
// Pack fp32 row-major (M x K) matrix into A-fragment-major bf16 tiles.
// ---------------------------------------------------------------------------
__global__ __launch_bounds__(256) void pack_a_kernel(
    const float* __restrict__ src, long ld, __bf16* __restrict__ dst,
    int Mtiles, int Ktiles, int M, int K) {
  long id = (long)blockIdx.x * 256 + threadIdx.x;
  long tile = id >> 5; int lane = id & 31;
  if (tile >= (long)Mtiles * Ktiles) return;
  int mt = (int)(tile / Ktiles), kb = (int)(tile % Ktiles);
  __bf16* d = dst + tile * 512 + lane * 16;
  int m = lane & 15, h = lane >> 4;
  #pragma unroll
  for (int e = 0; e < 16; ++e) {
    int v = e >> 1, ee = e & 1;
    int k = ((v >> 2) << 4) + ((v & 3) << 1) + ee + (h << 3);
    int rr = mt * 16 + m, cc = kb * 32 + k;
    float f = (rr < M && cc < K) ? src[(long)rr * ld + cc] : 0.f;
    d[e] = (__bf16)f;
  }
}

// ---------------------------------------------------------------------------
// Pack fp32 row-major (K x N) matrix into B-fragment-major bf16 tiles.
// ---------------------------------------------------------------------------
__global__ __launch_bounds__(256) void pack_b_kernel(
    const float* __restrict__ src, long ld, long srcBatch,
    __bf16* __restrict__ dst, long dstBatch,
    int Ktiles, int Ntiles, int K, int N) {
  long id = (long)blockIdx.x * 256 + threadIdx.x;
  long tile = id >> 5; int lane = id & 31;
  if (tile >= (long)Ktiles * Ntiles) return;
  const float* s = src + (long)blockIdx.z * srcBatch;
  __bf16* d = dst + (long)blockIdx.z * dstBatch + tile * 512 + lane * 16;
  int n = lane & 15, h = lane >> 4;
  #pragma unroll
  for (int e = 0; e < 16; ++e) {
    int k = (e & 7) + (h << 3) + ((e >> 3) << 4);
    int kk = (int)(tile / Ntiles) * 32 + k, nn = (int)(tile % Ntiles) * 16 + n;
    float f = (kk < K && nn < N) ? s[(long)kk * ld + nn] : 0.f;
    d[e] = (__bf16)f;
  }
}

// ---------------------------------------------------------------------------
// Generic WMMA GEMM, 2 M-tiles per wave: per k-step 2 A-frags + 1 shared
// B-frag feed two independent v_wmma accumulators.
// ---------------------------------------------------------------------------
__global__ __launch_bounds__(256) void gemm_kernel(
    const __bf16* __restrict__ A, const __bf16* __restrict__ B,
    float* __restrict__ out, const float* __restrict__ bias, int biasN,
    int Mtiles, int NtilesWork, int Ktiles, int NtilesPackB, int ntileOff,
    long outStride, int mode, long aBatch, long bBatch, long outBatch) {
  int lane = threadIdx.x & 31;
  long widx = (long)blockIdx.x * 8 + (threadIdx.x >> 5);
  long m2 = Mtiles >> 1;
  if (widx >= m2 * NtilesWork) return;            // wave-uniform
  int mt0 = 2 * (int)(widx / NtilesWork), nt = (int)(widx % NtilesWork);
  const __bf16* Ab = A + (long)blockIdx.z * aBatch + (long)mt0 * Ktiles * 512 + lane * 16;
  const __bf16* Bb = B + (long)blockIdx.z * bBatch + (long)(ntileOff + nt) * 512 + lane * 16;
  v8f acc0 = {0.f, 0.f, 0.f, 0.f, 0.f, 0.f, 0.f, 0.f};
  v8f acc1 = {0.f, 0.f, 0.f, 0.f, 0.f, 0.f, 0.f, 0.f};
  for (int kb = 0; kb < Ktiles; ++kb) {
    v16bf b  = *(const v16bf*)(Bb + (long)kb * NtilesPackB * 512);
    v16bf a0 = *(const v16bf*)(Ab + (long)kb * 512);
    v16bf a1 = *(const v16bf*)(Ab + (long)(Ktiles + kb) * 512);
    if (kb + 2 < Ktiles) {
      __builtin_prefetch(Bb + (long)(kb + 2) * NtilesPackB * 512, 0, 1);
    }
    acc0 = __builtin_amdgcn_wmma_f32_16x16x32_bf16(false, a0, false, b,
                                                   (short)0, acc0, false, false);
    acc1 = __builtin_amdgcn_wmma_f32_16x16x32_bf16(false, a1, false, b,
                                                   (short)0, acc1, false, false);
  }
  int n = lane & 15, h = lane >> 4;
  int col = nt * 16 + n;
  float bv = (bias && col < biasN) ? bias[col] : 0.f;
  float* ob = out + (long)blockIdx.z * outBatch;
  #pragma unroll
  for (int r = 0; r < 8; ++r) {
    int row0 = mt0 * 16 + r + (h << 3);
    float v0 = acc0[r] + bv;
    float v1 = acc1[r] + bv;
    if (mode == 1) { v0 = __expf(v0); v1 = __expf(v1); }
    ob[(long)row0 * outStride + col] = v0;
    ob[(long)(row0 + 16) * outStride + col] = v1;
  }
}

// ---------------------------------------------------------------------------
// k_t cumulative sum over time (att_k output).
// ---------------------------------------------------------------------------
__global__ void cumsum_kernel(const float* __restrict__ kinc,
                              const float* __restrict__ att_init,
                              float* __restrict__ attk) {
  int id = blockIdx.x * 256 + threadIdx.x;
  if (id >= 64 * 20) return;
  int b = id / 20, j = id % 20;
  float acc = att_init[b * 20 + j];
  for (int t = 0; t < 256; ++t) {
    acc += kinc[((long)t * 64 + b) * 32 + j];
    attk[((long)t * 64 + b) * 20 + j] = acc;
  }
}

// ---------------------------------------------------------------------------
// phi[t,b,c] = sum_j a*exp(-b*(k-c)^2); written directly as packed bf16
// A-fragments (per-batch 256x512 matrices, Mtiles=16, Ktiles=16).
// ---------------------------------------------------------------------------
__global__ __launch_bounds__(256) void phi_kernel(
    const float* __restrict__ Ae, const float* __restrict__ Be,
    const float* __restrict__ attk, __bf16* __restrict__ phiPack) {
  int row = blockIdx.x;                 // t*64 + b
  __shared__ float sa[20], sb[20], sk[20];
  int tid = threadIdx.x;
  if (tid < 20) {
    sa[tid] = Ae[(long)row * 32 + tid];
    sb[tid] = Be[(long)row * 32 + tid];
    sk[tid] = attk[(long)row * 20 + tid];
  }
  __syncthreads();
  int t = row >> 6, b = row & 63;
  for (int c = tid; c < 512; c += 256) {
    float s = 0.f;
    #pragma unroll
    for (int j = 0; j < 20; ++j) {
      float d = sk[j] - (float)c;
      s += sa[j] * __expf(-sb[j] * d * d);
    }
    long off = (long)b * 131072 + ((long)(t >> 4) * 16 + (c >> 5)) * 512
             + a_off(t & 15, c & 31);
    phiPack[off] = (__bf16)s;
  }
}

// ---------------------------------------------------------------------------
// x = ipre + att_w + bif, written directly as packed bf16 A-fragments.
// ---------------------------------------------------------------------------
__global__ void xfuse_kernel(const float* __restrict__ ipre,
                             const float* __restrict__ attw,
                             const float* __restrict__ bif,
                             __bf16* __restrict__ xpack) {
  long id = (long)blockIdx.x * 256 + threadIdx.x;
  if (id >= (long)16384 * 512) return;
  long row = id >> 9; int col = (int)(id & 511);
  float v = ipre[id] + attw[id] + bif[col];
  long off = ((row >> 4) * 16 + (col >> 5)) * 512 + a_off((int)(row & 15), col & 31);
  xpack[off] = (__bf16)v;
}

// ---------------------------------------------------------------------------
// Init h state and grid-barrier counter.
// ---------------------------------------------------------------------------
__global__ void seqinit_kernel(const float* __restrict__ gi,
                               float* __restrict__ hprev,
                               __bf16* __restrict__ hpack,
                               int* __restrict__ gcnt) {
  int id = blockIdx.x * 256 + threadIdx.x;
  if (id == 0) gcnt[0] = 0;
  if (id < 64 * 512) {
    float v = gi[id];
    hprev[id] = v;
    int row = id >> 9, col = id & 511;
    hpack[((row >> 4) * 16 + (col >> 5)) * 512 + a_off(row & 15, col & 31)] = (__bf16)v;
  }
}

// ---------------------------------------------------------------------------
// Sequential GRU scan.  4 persistent workgroups x 8 waves; each wave owns one
// 16-column N-tile; all four M-tiles accumulate simultaneously so each k-step
// needs 1 B-frag + 4 A-frags for 4 independent WMMAs.
//   phase1: reset = sigmoid(h @ Wur_r + G2);  sr = S * reset  -> packed bf16
//   phase2: h = tanh(sr @ U + S) + h + 1      (update gate cancels)
// ---------------------------------------------------------------------------
__global__ __launch_bounds__(256) void seq_kernel(
    const __bf16* __restrict__ WurrB, const __bf16* __restrict__ UB,
    const float* __restrict__ S, const float* __restrict__ G2,
    float* __restrict__ hprev, __bf16* __restrict__ hpack,
    __bf16* __restrict__ srpack, float* __restrict__ hid, int* gcnt) {
  const int tid = threadIdx.x, lane = tid & 31, wave = tid >> 5;
  const int nt = blockIdx.x * 8 + wave;  // 0..31 global N-tile
  const int n = lane & 15, h = lane >> 4;
  __shared__ int epoch;
  if (tid == 0) epoch = 0;
  __syncthreads();

  for (int t = 0; t < 256; ++t) {
    // ---- phase 1: reset gate & sr ----
    {
      v8f acc[4];
      #pragma unroll
      for (int mt = 0; mt < 4; ++mt) acc[mt] = (v8f){0.f,0.f,0.f,0.f,0.f,0.f,0.f,0.f};
      for (int kb = 0; kb < 16; ++kb) {
        v16bf b = *(const v16bf*)(WurrB + ((long)(kb * 32 + nt)) * 512 + lane * 16);
        #pragma unroll
        for (int mt = 0; mt < 4; ++mt) {
          v16bf a = *(const v16bf*)(hpack + ((long)(mt * 16 + kb)) * 512 + lane * 16);
          acc[mt] = __builtin_amdgcn_wmma_f32_16x16x32_bf16(false, a, false, b,
                                                            (short)0, acc[mt], false, false);
        }
      }
      #pragma unroll
      for (int mt = 0; mt < 4; ++mt) {
        #pragma unroll
        for (int r = 0; r < 8; ++r) {
          int row = mt * 16 + r + (h << 3);
          int col = nt * 16 + n;
          long gi = ((long)(t * 64 + row)) * 512 + col;
          float g = acc[mt][r] + G2[gi];
          float reset = 1.f / (1.f + __expf(-g));
          float sr = S[gi] * reset;
          srpack[((long)(row >> 4) * 16 + (col >> 5)) * 512 + a_off(row & 15, col & 31)]
              = (__bf16)sr;
        }
      }
    }
    // ---- grid barrier ----
    __threadfence(); __syncthreads();
    if (tid == 0) {
      int e = ++epoch;
      __hip_atomic_fetch_add(gcnt, 1, __ATOMIC_RELEASE, __HIP_MEMORY_SCOPE_AGENT);
      while (__hip_atomic_load(gcnt, __ATOMIC_ACQUIRE, __HIP_MEMORY_SCOPE_AGENT) < e * 4)
        __builtin_amdgcn_s_sleep(2);
    }
    __syncthreads();
    // ---- phase 2: candidate state & h update ----
    {
      v8f acc[4];
      #pragma unroll
      for (int mt = 0; mt < 4; ++mt) acc[mt] = (v8f){0.f,0.f,0.f,0.f,0.f,0.f,0.f,0.f};
      for (int kb = 0; kb < 16; ++kb) {
        v16bf b = *(const v16bf*)(UB + ((long)(kb * 32 + nt)) * 512 + lane * 16);
        #pragma unroll
        for (int mt = 0; mt < 4; ++mt) {
          v16bf a = *(const v16bf*)(srpack + ((long)(mt * 16 + kb)) * 512 + lane * 16);
          acc[mt] = __builtin_amdgcn_wmma_f32_16x16x32_bf16(false, a, false, b,
                                                            (short)0, acc[mt], false, false);
        }
      }
      #pragma unroll
      for (int mt = 0; mt < 4; ++mt) {
        #pragma unroll
        for (int r = 0; r < 8; ++r) {
          int row = mt * 16 + r + (h << 3);
          int col = nt * 16 + n;
          long gi = ((long)(t * 64 + row)) * 512 + col;
          float ns = tanhf(acc[mt][r] + S[gi]);
          float hn = ns + hprev[(long)row * 512 + col] + 1.0f;
          hid[gi] = hn;
          hprev[(long)row * 512 + col] = hn;
          hpack[((long)(row >> 4) * 16 + (col >> 5)) * 512 + a_off(row & 15, col & 31)]
              = (__bf16)hn;
        }
      }
    }
    __threadfence(); __syncthreads();
    if (tid == 0) {
      int e = ++epoch;
      __hip_atomic_fetch_add(gcnt, 1, __ATOMIC_RELEASE, __HIP_MEMORY_SCOPE_AGENT);
      while (__hip_atomic_load(gcnt, __ATOMIC_ACQUIRE, __HIP_MEMORY_SCOPE_AGENT) < e * 4)
        __builtin_amdgcn_s_sleep(2);
    }
    __syncthreads();
  }
}

// ---------------------------------------------------------------------------
extern "C" void kernel_launch(void* const* d_in, const int* in_sizes, int n_in,
                              void* d_out, int out_size, void* d_ws, size_t ws_size,
                              hipStream_t stream) {
  const float* c_inp   = (const float*)d_in[0];
  const float* inp     = (const float*)d_in[1];
  const float* gruinit = (const float*)d_in[2];
  const float* attinit = (const float*)d_in[3];
  const float* Wa  = (const float*)d_in[4];  const float* ba  = (const float*)d_in[5];
  const float* Wb  = (const float*)d_in[6];  const float* bbv = (const float*)d_in[7];
  const float* Wk  = (const float*)d_in[8];  const float* bk  = (const float*)d_in[9];
  const float* Wif = (const float*)d_in[10]; const float* bif = (const float*)d_in[11];
  const float* Wfork = (const float*)d_in[12]; const float* bfork = (const float*)d_in[13];
  const float* Wur = (const float*)d_in[14]; const float* U = (const float*)d_in[15];
  float* out = (float*)d_out;

  const long HID = 0;                       // hiddens (256,64,512)
  const long KOF = 8388608;                 // att_k   (256,64,20)
  const long WOF = 8716288;                 // att_w   (256,64,512)

  char* w = (char*)d_ws;
  size_t o = 0;
  auto alloc = [&](size_t bytes) { size_t r = o; o = (o + bytes + 255) & ~(size_t)255; return r; };
  __bf16* inpA   = (__bf16*)(w + alloc(16384ul * 512 * 2));
  __bf16* WaB    = (__bf16*)(w + alloc(16ul * 2 * 512 * 2));
  __bf16* WbB    = (__bf16*)(w + alloc(16ul * 2 * 512 * 2));
  __bf16* WkB    = (__bf16*)(w + alloc(16ul * 2 * 512 * 2));
  __bf16* WifB   = (__bf16*)(w + alloc(16ul * 32 * 512 * 2));
  __bf16* WforkB = (__bf16*)(w + alloc(16ul * 96 * 512 * 2));
  __bf16* WurrB  = (__bf16*)(w + alloc(16ul * 32 * 512 * 2));
  __bf16* UB     = (__bf16*)(w + alloc(16ul * 32 * 512 * 2));
  __bf16* cinpB  = (__bf16*)(w + alloc(64ul * 16 * 32 * 512 * 2));
  float*  Ae     = (float*) (w + alloc(16384ul * 32 * 4));
  float*  Be     = (float*) (w + alloc(16384ul * 32 * 4));
  float*  Kc     = (float*) (w + alloc(16384ul * 32 * 4));
  __bf16* phiP   = (__bf16*)(w + alloc(64ul * 16 * 16 * 512 * 2));
  float*  ipre   = (float*) (w + alloc(16384ul * 512 * 4));
  __bf16* xpack  = (__bf16*)(w + alloc(16384ul * 512 * 2));
  float*  Sbuf   = (float*) (w + alloc(16384ul * 512 * 4));
  float*  G2buf  = (float*) (w + alloc(16384ul * 512 * 4));
  float*  hprev  = (float*) (w + alloc(64ul * 512 * 4));
  __bf16* hpack  = (__bf16*)(w + alloc(64ul * 512 * 2));
  __bf16* srpack = (__bf16*)(w + alloc(64ul * 512 * 2));
  int*    gcnt   = (int*)   (w + alloc(256));

  // ---- pack activations & weights into WMMA fragments ----
  pack_a_kernel<<<2048, 256, 0, stream>>>(inp, 512, inpA, 1024, 16, 16384, 512);
  pack_b_kernel<<<4,   256, 0, stream>>>(Wa, 20, 0, WaB, 0, 16, 2, 512, 20);
  pack_b_kernel<<<4,   256, 0, stream>>>(Wb, 20, 0, WbB, 0, 16, 2, 512, 20);
  pack_b_kernel<<<4,   256, 0, stream>>>(Wk, 20, 0, WkB, 0, 16, 2, 512, 20);
  pack_b_kernel<<<64,  256, 0, stream>>>(Wif, 512, 0, WifB, 0, 16, 32, 512, 512);
  pack_b_kernel<<<192, 256, 0, stream>>>(Wfork, 1536, 0, WforkB, 0, 16, 96, 512, 1536);
  pack_b_kernel<<<64,  256, 0, stream>>>(Wur + 512, 1024, 0, WurrB, 0, 16, 32, 512, 512);
  pack_b_kernel<<<64,  256, 0, stream>>>(U, 512, 0, UB, 0, 16, 32, 512, 512);
  pack_b_kernel<<<dim3(64, 1, 64), 256, 0, stream>>>(
      c_inp, 64 * 512, 512, cinpB, 16 * 32 * 512, 16, 32, 512, 512);

  // ---- a,b,kinc = exp(inp @ W + b)  (fused exp epilogue) ----
  gemm_kernel<<<128, 256, 0, stream>>>(inpA, WaB, Ae, ba, 20, 1024, 2, 16, 2, 0, 32, 1, 0, 0, 0);
  gemm_kernel<<<128, 256, 0, stream>>>(inpA, WbB, Be, bbv, 20, 1024, 2, 16, 2, 0, 32, 1, 0, 0, 0);
  gemm_kernel<<<128, 256, 0, stream>>>(inpA, WkB, Kc, bk, 20, 1024, 2, 16, 2, 0, 32, 1, 0, 0, 0);

  // ---- k cumsum over time (att_k output) ----
  cumsum_kernel<<<5, 256, 0, stream>>>(Kc, attinit, out + KOF);

  // ---- phi, packed as per-batch A fragments ----
  phi_kernel<<<16384, 256, 0, stream>>>(Ae, Be, out + KOF, phiP);

  // ---- att_w = phi @ c_inp  (64 batched 256x512x512 GEMMs) ----
  gemm_kernel<<<dim3(32, 1, 64), 256, 0, stream>>>(
      phiP, cinpB, out + WOF, nullptr, 0, 16, 32, 16, 32, 0,
      64 * 512, 0, 131072, 16 * 32 * 512, 512);

  // ---- ipre = inp @ Wif ----
  gemm_kernel<<<2048, 256, 0, stream>>>(inpA, WifB, ipre, nullptr, 0,
                                        1024, 32, 16, 32, 0, 512, 0, 0, 0, 0);
  // ---- x = att_w + ipre + bif, packed ----
  xfuse_kernel<<<32768, 256, 0, stream>>>(ipre, out + WOF, bif, xpack);

  // ---- S = x @ Wfork[:, :512] + bfork[:512]   (state_inp)
  //      G2 = x @ Wfork[:,1024:] + bfork[1024:] (reset-gate input) ----
  gemm_kernel<<<2048, 256, 0, stream>>>(xpack, WforkB, Sbuf, bfork, 512,
                                        1024, 32, 16, 96, 0, 512, 0, 0, 0, 0);
  gemm_kernel<<<2048, 256, 0, stream>>>(xpack, WforkB, G2buf, bfork + 1024, 512,
                                        1024, 32, 16, 96, 64, 512, 0, 0, 0, 0);

  // ---- sequential GRU scan (persistent, 4 WGs, atomic grid barrier) ----
  seqinit_kernel<<<128, 256, 0, stream>>>(gruinit, hprev, hpack, gcnt);
  seq_kernel<<<4, 256, 0, stream>>>(WurrB, UB, Sbuf, G2buf, hprev, hpack,
                                    srpack, out + HID, gcnt);
  (void)in_sizes; (void)n_in; (void)out_size; (void)ws_size;
}
